// ActorSpine_20169166422128
// MI455X (gfx1250) — matrix operand order repaired and checked by reading, not compiled, not run on hardware
//
#include <hip/hip_runtime.h>
#include <hip/hip_bf16.h>

// ---------------------------------------------------------------------------
// MI455X (gfx1250). Compute-bound (~245 GFLOP vs ~10us of HBM traffic at
// 23.3 TB/s) -> all three GEMMs on v_wmma_f32_16x16x32_f16 (wave32, f32 acc),
// CDNA5 async global->LDS copies (ASYNCcnt) + double-buffered LDS.
// Wave tile 64x64 => 16 WMMA per 32-K slice per wave (dense compute loop).
// ---------------------------------------------------------------------------

typedef __attribute__((ext_vector_type(16))) _Float16 v16h;
typedef __attribute__((ext_vector_type(8)))  float    v8f;

#define BATCH      16384
#define STATE_DIM  128
#define ENC_K      10
#define SPINE_K    1280          // STATE_DIM * ENC_K
#define H0         2048
#define H1         2048
#define ACT_DIM    32
#define DEC_K      10
#define OUT3       320           // ACT_DIM * DEC_K

// -------------------- CDNA5 async copy primitives --------------------------
__device__ __forceinline__ void async_copy_b128(unsigned lds_off, const void* gaddr)
{
    // GLOBAL_LOAD_ASYNC_TO_LDS_B128: per-lane 16B global -> LDS, ASYNCcnt.
    asm volatile("global_load_async_to_lds_b128 %0, %1, off"
                 :: "v"(lds_off), "v"(gaddr) : "memory");
}

__device__ __forceinline__ void wait_async0()
{
#if __has_builtin(__builtin_amdgcn_s_wait_asynccnt)
    __builtin_amdgcn_s_wait_asynccnt(0);
#else
    asm volatile("s_wait_asynccnt 0x0" ::: "memory");
#endif
}

// ------------------------------ encoder ------------------------------------
__global__ __launch_bounds__(256)
void enc_kernel(const float* __restrict__ state,
                const float* __restrict__ mean,
                const float* __restrict__ stdv,
                _Float16* __restrict__ spine)
{
    int idx = blockIdx.x * 256 + threadIdx.x;      // over BATCH*STATE_DIM
    int d   = idx & (STATE_DIM - 1);
    float s = state[idx];
    _Float16* o = spine + (size_t)(idx >> 7) * SPINE_K + d * ENC_K;
#pragma unroll
    for (int j = 0; j < ENC_K; ++j) {
        float x = (s - mean[d * ENC_K + j]) / stdv[d * ENC_K + j];
        o[j] = (_Float16)(1.0f / (1.0f + expf(-x)));
    }
}

// -------------------------- fp32 -> f16 convert ----------------------------
__global__ __launch_bounds__(256)
void cvt_kernel(const float* __restrict__ src, _Float16* __restrict__ dst, int n)
{
    for (int i = blockIdx.x * 256 + threadIdx.x; i < n; i += gridDim.x * 256)
        dst[i] = (_Float16)src[i];
}

// ------------------------------ WMMA GEMM ----------------------------------
// C[M,N] = A[M,K] * W[N,K]^T + bias ; optional ReLU ; f16 or f32 output.
// Block: 256 threads (8 waves). Block tile BM x BN, wave tile WM x WN.
// LDS row stride = 40 halves (80B): 16B-aligned b128 + conflict-free banks.
// K staged 32 at a time via async global->LDS, double buffered.
template<int BM, int BN, int WM, int WN, bool RELU, bool F16OUT>
__global__ __launch_bounds__(256)
void gemm_wmma(const _Float16* __restrict__ A,
               const _Float16* __restrict__ W,
               const float*    __restrict__ bias,
               void*           __restrict__ out,
               int M, int N, int K)
{
    constexpr int LS  = 40;            // halves per 32-wide K slice (padded)
    constexpr int LSB = LS * 2;        // bytes per row
    constexpr int WAVES_M = BM / WM, WAVES_N = BN / WN;
    static_assert(WAVES_M * WAVES_N == 8, "8 waves per block");
    constexpr int MT = WM / 16, NT = WN / 16;
    constexpr int BUFH = (BM + BN) * LS;       // halves per buffer
    constexpr int AC = (BM * 4) / 256;         // b128 chunks per thread (A)
    constexpr int BC = (BN * 4) / 256;         // b128 chunks per thread (B)
    static_assert(AC >= 1 && BC >= 1, "");

    __shared__ _Float16 smem[2 * BUFH];

    const int t    = threadIdx.x;
    const int lane = t & 31, wid = t >> 5;
    const int wm = wid % WAVES_M, wn = wid / WAVES_M;
    const int m0 = blockIdx.x * BM, n0 = blockIdx.y * BN;
    const int lane15 = lane & 15, laneHi = lane >> 4;

    // per-thread staging descriptors (fixed across K loop)
    const unsigned ldsBase = (unsigned)(size_t)(void*)smem; // flat low 32 = LDS offset
    const char* gA[AC]; unsigned la[AC];
    const char* gB[BC]; unsigned lb[BC];
#pragma unroll
    for (int c = 0; c < AC; ++c) {
        int idx = t + c * 256, row = idx >> 2, seg = (idx & 3) * 16;
        gA[c] = (const char*)(A + (size_t)(m0 + row) * K) + seg;
        la[c] = row * LSB + seg;
    }
#pragma unroll
    for (int c = 0; c < BC; ++c) {
        int idx = t + c * 256, row = idx >> 2, seg = (idx & 3) * 16;
        gB[c] = (const char*)(W + (size_t)(n0 + row) * K) + seg;
        lb[c] = BM * LSB + row * LSB + seg;
    }

    auto stage = [&](int kk, int p) {
        unsigned boff = ldsBase + (unsigned)p * (BUFH * 2);
#pragma unroll
        for (int c = 0; c < AC; ++c)
            async_copy_b128(boff + la[c], gA[c] + (size_t)kk * 2);
#pragma unroll
        for (int c = 0; c < BC; ++c)
            async_copy_b128(boff + lb[c], gB[c] + (size_t)kk * 2);
    };

    v8f acc[MT][NT];
    const v8f vzero = {0.f, 0.f, 0.f, 0.f, 0.f, 0.f, 0.f, 0.f};
#pragma unroll
    for (int i = 0; i < MT; ++i)
#pragma unroll
        for (int j = 0; j < NT; ++j) acc[i][j] = vzero;

    stage(0, 0);
    int p = 0;
    for (int k0 = 0; k0 < K; k0 += 32) {
        wait_async0();          // this wave's async writes have landed in LDS
        __syncthreads();        // every wave's writes landed; prev compute done
        if (k0 + 32 < K) stage(k0 + 32, p ^ 1);

        const char* bufA = (const char*)smem + (size_t)p * (BUFH * 2);
        const char* bufB = bufA + BM * LSB;

        // A fragments: ISA 16-bit A layout — lanes 0-15: K0-7 & K16-23,
        // lanes 16-31: K8-15 & K24-31, of row M = lane&15.
        v16h afrag[MT];
#pragma unroll
        for (int i = 0; i < MT; ++i) {
            int row = wm * WM + i * 16 + lane15;
            int off = row * LSB + laneHi * 16;
            union { v16h v; uint4 q[2]; } u;
            u.q[0] = *(const uint4*)(bufA + off);
            u.q[1] = *(const uint4*)(bufA + off + 32);
            afrag[i] = u.v;
        }
        // B fragments: lanes 0-15 hold K0-7 & K16-23, lanes 16-31 hold
        // K8-15 & K24-31 of column N = lane&15 (same k order as A).
        v16h bfrag[NT];
#pragma unroll
        for (int j = 0; j < NT; ++j) {
            int col = wn * WN + j * 16 + lane15;
            int off = col * LSB + laneHi * 16;
            union { v16h v; uint4 q[2]; } u;
            u.q[0] = *(const uint4*)(bufB + off);
            u.q[1] = *(const uint4*)(bufB + off + 32);
            bfrag[j] = u.v;
        }
#pragma unroll
        for (int i = 0; i < MT; ++i)
#pragma unroll
            for (int j = 0; j < NT; ++j)
                acc[i][j] = __builtin_amdgcn_wmma_f32_16x16x32_f16(
                    false, afrag[i], false, bfrag[j],
                    (short)0, acc[i][j], false, false);
        p ^= 1;
    }

    // epilogue: C layout — VGPR r holds row (r + 8*laneHi), col = lane&15
#pragma unroll
    for (int i = 0; i < MT; ++i)
#pragma unroll
        for (int j = 0; j < NT; ++j) {
            int tn = n0 + wn * WN + j * 16 + lane15;
            int tm = m0 + wm * WM + i * 16 + laneHi * 8;
            float bv = bias[tn];
#pragma unroll
            for (int r = 0; r < 8; ++r) {
                float v = acc[i][j][r] + bv;
                if (RELU) v = v > 0.f ? v : 0.f;
                size_t o = (size_t)(tm + r) * N + tn;
                if (F16OUT) ((_Float16*)out)[o] = (_Float16)v;
                else        ((float*)out)[o]    = v;
            }
        }
}

// ------------------------------ decoder ------------------------------------
__global__ __launch_bounds__(256)
void dec_kernel(const float* __restrict__ raw,
                const float* __restrict__ Wd,
                const float* __restrict__ bd,
                float* __restrict__ out)
{
    int idx = blockIdx.x * 256 + threadIdx.x;      // over BATCH*ACT_DIM
    int a   = idx & (ACT_DIM - 1);
    const float* r = raw + (size_t)(idx >> 5) * OUT3 + a * DEC_K;
    float acc = bd[a];
#pragma unroll
    for (int k = 0; k < DEC_K; ++k) acc += r[k] * Wd[a * DEC_K + k];
    out[idx] = tanhf(acc);
}

// ------------------------------ launcher -----------------------------------
extern "C" void kernel_launch(void* const* d_in, const int* in_sizes, int n_in,
                              void* d_out, int out_size, void* d_ws, size_t ws_size,
                              hipStream_t stream)
{
    const float* state    = (const float*)d_in[0];
    const float* mean_enc = (const float*)d_in[1];
    const float* std_enc  = (const float*)d_in[2];
    const float* W1 = (const float*)d_in[3];
    const float* b1 = (const float*)d_in[4];
    const float* W2 = (const float*)d_in[5];
    const float* b2 = (const float*)d_in[6];
    const float* W3 = (const float*)d_in[7];
    const float* b3 = (const float*)d_in[8];
    const float* Wd = (const float*)d_in[9];
    const float* bd = (const float*)d_in[10];
    float* out = (float*)d_out;
    char*  ws  = (char*)d_ws;

    // workspace layout (with lifetime-based aliasing), ~137 MB total
    constexpr size_t SZ_W2H   = (size_t)H1 * H0 * 2;        //  8 MB
    constexpr size_t SZ_W3H   = (size_t)OUT3 * H1 * 2;      //  1.25 MB
    constexpr size_t SZ_ACT1  = (size_t)BATCH * H0 * 2;     // 64 MB
    constexpr size_t SZ_SPINE = (size_t)BATCH * SPINE_K * 2;// 40 MB

    constexpr size_t OFF_W2H   = 0;
    constexpr size_t OFF_W3H   = OFF_W2H + SZ_W2H;
    constexpr size_t OFF_ACT1  = OFF_W3H + SZ_W3H;
    constexpr size_t OFF_SPINE = OFF_ACT1 + SZ_ACT1;
    constexpr size_t OFF_W1H   = OFF_SPINE + SZ_SPINE;
    constexpr size_t OFF_ACT2  = OFF_SPINE;   // aliases spine+W1H (dead after GEMM1)
    constexpr size_t OFF_RAW   = OFF_ACT1;    // aliases act1 (dead after GEMM2)

    _Float16* W1H    = (_Float16*)(ws + OFF_W1H);
    _Float16* W2H    = (_Float16*)(ws + OFF_W2H);
    _Float16* W3H    = (_Float16*)(ws + OFF_W3H);
    _Float16* spineH = (_Float16*)(ws + OFF_SPINE);
    _Float16* act1H  = (_Float16*)(ws + OFF_ACT1);
    _Float16* act2H  = (_Float16*)(ws + OFF_ACT2);
    float*    rawF   = (float*)   (ws + OFF_RAW);

    // 1) weight conversion fp32 -> f16
    cvt_kernel<<<2048, 256, 0, stream>>>(W1, W1H, H0 * SPINE_K);
    cvt_kernel<<<2048, 256, 0, stream>>>(W2, W2H, H1 * H0);
    cvt_kernel<<<512,  256, 0, stream>>>(W3, W3H, OUT3 * H1);

    // 2) population-coding encoder -> spine (f16)
    enc_kernel<<<(BATCH * STATE_DIM) / 256, 256, 0, stream>>>(
        state, mean_enc, std_enc, spineH);

    // 3) GEMM1: [16384,1280] x [2048,1280]^T, ReLU, f16 out  (wave tile 64x64)
    gemm_wmma<256, 128, 64, 64, true, true>
        <<<dim3(BATCH / 256, H0 / 128), 256, 0, stream>>>(
            spineH, W1H, b1, act1H, BATCH, H0, SPINE_K);

    // 4) GEMM2: [16384,2048] x [2048,2048]^T, ReLU, f16 out  (wave tile 64x64)
    gemm_wmma<256, 128, 64, 64, true, true>
        <<<dim3(BATCH / 256, H1 / 128), 256, 0, stream>>>(
            act1H, W2H, b2, act2H, BATCH, H1, H0);

    // 5) GEMM3: [16384,2048] x [320,2048]^T, no ReLU, f32 out
    gemm_wmma<128, 64, 64, 16, false, false>
        <<<dim3(BATCH / 128, OUT3 / 64), 256, 0, stream>>>(
            act2H, W3H, b3, rawF, BATCH, OUT3, H1);

    // 6) per-channel decoder dot + tanh
    dec_kernel<<<(BATCH * ACT_DIM) / 256, 256, 0, stream>>>(rawF, Wd, bd, out);
}
